// QKVAttentionG1_18021682775002
// MI455X (gfx1250) — matrix operand order repaired and checked by reading, not compiled, not run on hardware
//
#include <hip/hip_runtime.h>

typedef __attribute__((ext_vector_type(16))) __bf16 v16bf;
typedef __attribute__((ext_vector_type(8)))  __bf16 v8bf;
typedef __attribute__((ext_vector_type(4)))  __bf16 v4bf;
typedef __attribute__((ext_vector_type(8)))  float  v8f;

// ---------------------------------------------------------------------------
// Fragment loaders (CDNA5 ISA 7.12.2, wave32). Global loaders take
// (kernarg base, element offset) so addrspace inference emits global_load_b128
// (flat_load would tie up DScnt and the LDS pipe used for the P transpose).
//  A (16x32 bf16): lanes 0-15 = rows M0-15 holding K {0..7,16..23};
//                  lanes 16-31 = same rows holding K {8..15,24..31}.
//  B (32x16 bf16): lanes 0-15 = cols N0-15 holding K {0..7,16..23};
//                  lanes 16-31 = same cols holding K {8..15,24..31}.
// ---------------------------------------------------------------------------
static __device__ __forceinline__ v16bf load_a_g(const __bf16* __restrict__ base,
                                                 size_t off, int half) {
    const v8bf lo = *(const v8bf*)(base + off + half * 8);
    const v8bf hi = *(const v8bf*)(base + off + half * 8 + 16);
    v16bf r;
#pragma unroll
    for (int i = 0; i < 8; ++i) { r[i] = lo[i]; r[i + 8] = hi[i]; }
    return r;
}

static __device__ __forceinline__ v16bf load_b_g(const __bf16* __restrict__ base,
                                                 size_t off, int half) {
    const v8bf lo = *(const v8bf*)(base + off + half * 8);
    const v8bf hi = *(const v8bf*)(base + off + half * 8 + 16);
    v16bf r;
#pragma unroll
    for (int i = 0; i < 8; ++i) { r[i] = lo[i]; r[i + 8] = hi[i]; }
    return r;
}

// LDS variant (P tile): pointer rooted in __shared__ -> ds_load
static __device__ __forceinline__ v16bf load_a_lds(const __bf16* row, int kc, int half) {
    const v8bf lo = *(const v8bf*)(row + kc + half * 8);
    const v8bf hi = *(const v8bf*)(row + kc + half * 8 + 16);
    v16bf r;
#pragma unroll
    for (int i = 0; i < 8; ++i) { r[i] = lo[i]; r[i + 8] = hi[i]; }
    return r;
}

static __device__ __forceinline__ v8f wmma_bf16(v16bf a, v16bf b, v8f c) {
    return __builtin_amdgcn_wmma_f32_16x16x32_bf16(false, a, false, b, (short)0, c, false, false);
}

// Row-of-16 xor-reduction max. Prefer v_permlane16_b32 (pure VALU; keeps the
// LDS pipe free) over ds_bpermute-based __shfl_xor.
static __device__ __forceinline__ float xor16_max(float v) {
#if __has_builtin(__builtin_amdgcn_permlane16)
    int x = __float_as_int(v);
    // sel nibbles encode lane^1 / ^2 / ^4 / ^8 within each row of 16
    x = __float_as_int(v = fmaxf(v, __int_as_float(
        __builtin_amdgcn_permlane16(x, x, 0x67452301u, 0xEFCDAB89u, false, false))));
    x = __float_as_int(v = fmaxf(v, __int_as_float(
        __builtin_amdgcn_permlane16(x, x, 0x54761032u, 0xDCFE98BAu, false, false))));
    x = __float_as_int(v = fmaxf(v, __int_as_float(
        __builtin_amdgcn_permlane16(x, x, 0x32107654u, 0xBA98FEDCu, false, false))));
    v = fmaxf(v, __int_as_float(
        __builtin_amdgcn_permlane16(x, x, 0xFEDCBA98u, 0x76543210u, false, false)));
    return v;
#else
    v = fmaxf(v, __shfl_xor(v, 1, 16));
    v = fmaxf(v, __shfl_xor(v, 2, 16));
    v = fmaxf(v, __shfl_xor(v, 4, 16));
    v = fmaxf(v, __shfl_xor(v, 8, 16));
    return v;
#endif
}

static __device__ __forceinline__ float gelu_exact(float x) {
    return 0.5f * x * (1.0f + erff(x * 0.70710678118654752f));
}

// ---------------------------------------------------------------------------
// fp32 -> bf16 conversion (grid-stride)
// ---------------------------------------------------------------------------
__global__ void cast_bf16_kernel(const float* __restrict__ src, __bf16* __restrict__ dst, int n) {
    int i = blockIdx.x * blockDim.x + threadIdx.x;
    int stride = gridDim.x * blockDim.x;
    for (; i < n; i += stride) dst[i] = (__bf16)src[i];
}

// ---------------------------------------------------------------------------
// Fused projections. M=4096 (b*s), N=K=1024. y = x @ W^T + b.
// Row-major W is directly the B operand (B[k][n] = W[n][k]).
// TASK 0: 16x32 of BOTH q and g -> gelu(q*g) bf16 [b,h,s,hd]
// TASK 1: 16x64 of k            -> bf16 [b,h,s,hd]
// TASK 2: 16x64 of v            -> bf16 [b,h,hd,s'], s' = column-permuted
//         within each 64-block (s' block pos c holds k = 16*(c&3)+(c>>2)),
//         matching the packed P layout in the attention kernel.
// ---------------------------------------------------------------------------
template <int TASK>
__global__ __launch_bounds__(256) void proj_qkvg_kernel(
    const __bf16* __restrict__ xb,   // [4096,1024]
    const __bf16* __restrict__ wb,   // [4][1024][1024]  (q,k,v,g)
    const float* __restrict__ bias0, // TASK0: bq, TASK1: bk, TASK2: bv
    const float* __restrict__ bias1, // TASK0: bg, else unused
    __bf16* __restrict__ dst)
{
    const int lane = threadIdx.x & 31;
    const int wv   = threadIdx.x >> 5;
    const int half = lane >> 4;
    const int lm   = lane & 15;

    const int id = blockIdx.x * 8 + wv;
    const int NT = (TASK == 0) ? 2 : 4;                      // 16-wide N tiles
    const int m0 = (TASK == 0) ? (id >> 5) * 16 : (id >> 4) * 16;
    const int n0 = (TASK == 0) ? (id & 31) * 32 : (id & 15) * 64;

    const size_t aoff = (size_t)(m0 + lm) * 1024;
    const size_t wsel = (size_t)((TASK == 0) ? 0 : (TASK == 1 ? 1 : 2)) * 1048576u;
    size_t w0off[4], w1off[4];
#pragma unroll
    for (int t = 0; t < NT; ++t) {
        w0off[t] = wsel + (size_t)(n0 + t * 16 + lm) * 1024;
        w1off[t] = 3u * 1048576u + (size_t)(n0 + t * 16 + lm) * 1024;
    }

    v8f c0[4] = {};   // q / k / v
    v8f c1[2] = {};   // g (TASK 0 only)

#pragma unroll 2
    for (int kc = 0; kc < 1024; kc += 32) {
        const v16bf a = load_a_g(xb, aoff + kc, half);
#pragma unroll
        for (int t = 0; t < NT; ++t) {
            c0[t] = wmma_bf16(a, load_b_g(wb, w0off[t] + kc, half), c0[t]);
            if (TASK == 0)
                c1[t] = wmma_bf16(a, load_b_g(wb, w1off[t] + kc, half), c1[t]);
        }
    }

#pragma unroll
    for (int t = 0; t < NT; ++t) {
        const int n  = n0 + t * 16 + lm;
        const int h  = n >> 6;
        const int hd = n & 63;
        const float b0v = bias0[n];
        const float b1v = (TASK == 0) ? bias1[n] : 0.0f;
#pragma unroll
        for (int r = 0; r < 8; ++r) {
            const int m  = m0 + r + 8 * half;    // 0..4095
            const int bi = m >> 11;
            const int s  = m & 2047;
            if (TASK == 0) {
                float q = c0[t][r] + b0v;
                float g = c1[t][r] + b1v;
                dst[(((size_t)bi * 16 + h) * 2048 + s) * 64 + hd] = (__bf16)gelu_exact(q * g);
            } else if (TASK == 1) {
                dst[(((size_t)bi * 16 + h) * 2048 + s) * 64 + hd] = (__bf16)(c0[t][r] + b0v);
            } else {
                const int o  = s & 63;                        // pos in 64-block
                const int sp = (s & ~63) | (((o & 15) << 2) | (o >> 4));
                dst[(((size_t)bi * 16 + h) * 64 + hd) * 2048 + sp] = (__bf16)(c0[t][r] + b0v);
            }
        }
    }
}

// ---------------------------------------------------------------------------
// Flash attention, no 1/sqrt(d) scale (per reference).
// Grid (S/128, B*H); block 256 = 8 waves; wave owns 16 q-rows.
// Per 64-column k-chunk: 8 score WMMA, permlane16 row-max, exp, packed P
// store (one ds_store_b64 per row), 2 ones-WMMA for the row sums (replaces
// all cross-lane sum shuffles), 8 PV WMMA. P goes through a wave-private LDS
// tile; wave DS ops are in-order so no barrier is needed.
// ---------------------------------------------------------------------------
#define PSTR 72   // LDS row stride in halves: 64 cols + pad, multiple of 8

__global__ __launch_bounds__(256) void attn_kernel(
    const __bf16* __restrict__ qg,   // [b,h,s,hd]
    const __bf16* __restrict__ kbuf, // [b,h,s,hd]
    const __bf16* __restrict__ vP,   // [b,h,hd,s'] column-permuted
    float* __restrict__ out)         // [B,S,D]
{
    __shared__ __bf16 lds[8 * 16 * PSTR];

    const int lane = threadIdx.x & 31;
    const int wv   = threadIdx.x >> 5;
    const int half = lane >> 4;
    const int lm   = lane & 15;

    const int bh = blockIdx.y;
    const int b  = bh >> 4;
    const int h  = bh & 15;
    const int m0 = blockIdx.x * 128 + wv * 16;

    const size_t bhS  = ((size_t)b * 16 + h) * 2048;
    const size_t vbhD = ((size_t)b * 16 + h) * 64;

    const v16bf a0 = load_a_g(qg, (bhS + m0 + lm) * 64,      half);
    const v16bf a1 = load_a_g(qg, (bhS + m0 + lm) * 64 + 32, half);

    v16bf ones;
#pragma unroll
    for (int i = 0; i < 16; ++i) ones[i] = (__bf16)1.0f;

    float m_st[8], l_st[8];
    v8f oacc[4] = {};
#pragma unroll
    for (int r = 0; r < 8; ++r) { m_st[r] = -1.0e30f; l_st[r] = 0.0f; }

    __bf16* pl = &lds[wv * 16 * PSTR];

    for (int kc = 0; kc < 2048; kc += 64) {
        // --- scores: 4 independent 16x16 tiles over 64 k-columns ---
        v8f s[4] = {};
#pragma unroll
        for (int t = 0; t < 4; ++t) {
            const size_t koff = (bhS + kc + t * 16 + lm) * 64;
            s[t] = wmma_bf16(a0, load_b_g(kbuf, koff, half),      s[t]);
            s[t] = wmma_bf16(a1, load_b_g(kbuf, koff + 32, half), s[t]);
            if (kc + 64 < 2048)                                   // next chunk
                __builtin_prefetch(kbuf + koff + 64 * 64, 0, 3);  // -> WGP$
        }

        // --- online softmax: VALU-only row max, packed P store ---
        float scr[8];
#pragma unroll
        for (int r = 0; r < 8; ++r) {
            float v = fmaxf(fmaxf(s[0][r], s[1][r]), fmaxf(s[2][r], s[3][r]));
            v = xor16_max(v);
            const float mn = fmaxf(m_st[r], v);
            scr[r]  = __expf(m_st[r] - mn);
            m_st[r] = mn;
            v4bf pk;
#pragma unroll
            for (int t = 0; t < 4; ++t) pk[t] = (__bf16)__expf(s[t][r] - mn);
            // packed column layout: LDS col c=4*lm+t holds k = kc+16*t+lm
            *(v4bf*)(pl + (r + 8 * half) * PSTR + 4 * lm) = pk;
        }
#pragma unroll
        for (int t = 0; t < 4; ++t)
#pragma unroll
            for (int r = 0; r < 8; ++r) oacc[t][r] *= scr[r];

        // --- P A-frags from LDS ---
        const v16bf pf0 = load_a_lds(pl + lm * PSTR, 0,  half);
        const v16bf pf1 = load_a_lds(pl + lm * PSTR, 32, half);

        // --- row sums via ones-WMMA (every column of lsum = rowsum(P)) ---
        v8f lsum = {};
        lsum = wmma_bf16(pf0, ones, lsum);
        lsum = wmma_bf16(pf1, ones, lsum);
#pragma unroll
        for (int r = 0; r < 8; ++r) l_st[r] = l_st[r] * scr[r] + lsum[r];

        // --- PV over 4 hd-tiles (vP uses the same column permutation) ---
#pragma unroll
        for (int t = 0; t < 4; ++t) {
            const size_t voff = (vbhD + t * 16 + lm) * 2048 + kc;
            oacc[t] = wmma_bf16(pf0, load_b_g(vP, voff, half),      oacc[t]);
            oacc[t] = wmma_bf16(pf1, load_b_g(vP, voff + 32, half), oacc[t]);
        }
    }

    // --- normalize + store fp32 out[b][s][h*64+d] ---
#pragma unroll
    for (int t = 0; t < 4; ++t) {
#pragma unroll
        for (int r = 0; r < 8; ++r) {
            const int s = m0 + r + 8 * half;
            out[((size_t)b * 2048 + s) * 1024 + h * 64 + t * 16 + lm] = oacc[t][r] / l_st[r];
        }
    }
}

// ---------------------------------------------------------------------------
// Launch
// ---------------------------------------------------------------------------
extern "C" void kernel_launch(void* const* d_in, const int* in_sizes, int n_in,
                              void* d_out, int out_size, void* d_ws, size_t ws_size,
                              hipStream_t stream) {
    (void)in_sizes; (void)n_in; (void)out_size; (void)ws_size;

    const float* x  = (const float*)d_in[0];
    const float* Wq = (const float*)d_in[1]; const float* bq = (const float*)d_in[2];
    const float* Wk = (const float*)d_in[3]; const float* bk = (const float*)d_in[4];
    const float* Wv = (const float*)d_in[5]; const float* bv = (const float*)d_in[6];
    const float* Wg = (const float*)d_in[7]; const float* bg = (const float*)d_in[8];

    // workspace layout (bf16 elements)
    __bf16* base = (__bf16*)d_ws;
    __bf16* xb   = base;                 // 4096*1024
    __bf16* wb   = xb  + 4194304;        // 4*1024*1024 (q,k,v,g)
    __bf16* qg   = wb  + 4194304;        // [b,h,s,hd]
    __bf16* kbuf = qg  + 4194304;        // [b,h,s,hd]
    __bf16* vP   = kbuf + 4194304;       // [b,h,hd,s'] permuted

    cast_bf16_kernel<<<2048, 256, 0, stream>>>(x,  xb,             4194304);
    cast_bf16_kernel<<<512,  256, 0, stream>>>(Wq, wb + 0*1048576, 1048576);
    cast_bf16_kernel<<<512,  256, 0, stream>>>(Wk, wb + 1*1048576, 1048576);
    cast_bf16_kernel<<<512,  256, 0, stream>>>(Wv, wb + 2*1048576, 1048576);
    cast_bf16_kernel<<<512,  256, 0, stream>>>(Wg, wb + 3*1048576, 1048576);

    proj_qkvg_kernel<0><<<1024, 256, 0, stream>>>(xb, wb, bq, bg, qg);
    proj_qkvg_kernel<1><<<512,  256, 0, stream>>>(xb, wb, bk, nullptr, kbuf);
    proj_qkvg_kernel<2><<<512,  256, 0, stream>>>(xb, wb, bv, nullptr, vP);

    attn_kernel<<<dim3(16, 32), 256, 0, stream>>>(qg, kbuf, vP, (float*)d_out);
}
